// _GNNSimulator_76321568850178
// MI455X (gfx1250) — matrix operand order repaired and checked by reading, not compiled, not run on hardware
//
#include <hip/hip_runtime.h>
#include <cstdint>

// ---------------------------------------------------------------------------
// Types for CDNA5 WMMA (gfx1250, wave32)
// ---------------------------------------------------------------------------
typedef __attribute__((ext_vector_type(16))) _Float16 v16h;
typedef __attribute__((ext_vector_type(8)))  float    v8f;

union FragH {
    uint4 u[2];
    v16h  v;
};

// LDS row stride (halves) for the weight tile, padded vs bank conflicts.
// 136*2 = 272 bytes: every 8-half (16B) segment stays 16B aligned.
#define WLDS_STRIDE 136

__device__ __forceinline__ void atomAddF(float* p, float v) {
    unsafeAtomicAdd(p, v);   // lowers to global_atomic_add_f32 on gfx1250
}

// ---------------------------------------------------------------------------
// Generic fused MLP-layer GEMM:  Y[rows,Nout] = act(A[rows,Kpad] * W + b)
//   A  : f16 row-major, zero padded to Kpad (multiple of 32)
//   Wt : f16, pre-transposed/padded [128 (n)][Kpad (k)] — staged to LDS by
//        global_load_async_to_lds_b128 (ASYNCcnt path, no VGPR round-trip)
// One block = 256 threads = 8 waves; each wave computes 16 rows x 128 cols.
// MODE 0: ReLU, write f16
// MODE 1: no act, write f16 + f32 (encoder output)
// MODE 2: edge-final: m=acc+b; ef_f32 += m; ef_f16 = cvt; atomicAdd aggr[dst]
// MODE 3: node-final: nf_f32 += acc+b; nf_f16 = cvt
// MODE 4: action head: write f32 dout[row*3+col], col<Nout(=3)
// ---------------------------------------------------------------------------
template <int NTILES, int MODE>
__launch_bounds__(256)
__global__ void mlp_gemm(const _Float16* __restrict__ A, int lda, int rows,
                         const _Float16* __restrict__ Wt,
                         const float* __restrict__ bias,
                         int Kpad, int Nout,
                         _Float16* __restrict__ outH,
                         float* __restrict__ outF,
                         float* __restrict__ aggr,
                         const int* __restrict__ dstIdx,
                         float* __restrict__ dout) {
    __shared__ __align__(16) _Float16 Wlds[128 * WLDS_STRIDE];
    __shared__ float blds[128];

    const int tid    = threadIdx.x;
    const int lane   = tid & 31;
    const int wave   = tid >> 5;
    const int laneHi = (lane >> 4) & 1;           // lanes 16..31
    const int lm     = lane & 15;
    const int row0   = blockIdx.x * 128 + wave * 16;

    if (tid < 128) blds[tid] = (tid < Nout) ? bias[tid] : 0.0f;

    v8f acc[NTILES];
#pragma unroll
    for (int t = 0; t < NTILES; ++t)
#pragma unroll
        for (int r = 0; r < 8; ++r) acc[t][r] = 0.0f;

    int rowA = row0 + lm;
    if (rowA >= rows) rowA = rows - 1;            // clamp; stores are guarded
    const uint4* __restrict__ Arow = (const uint4*)(A + (size_t)rowA * (size_t)lda);

    const uint32_t ldsBase = (uint32_t)(uintptr_t)(&Wlds[0]);

    for (int kc = 0; kc < Kpad; kc += 128) {
        const int ck = (Kpad - kc) < 128 ? (Kpad - kc) : 128;

        // ---- async DMA W tile [128 n][ck k] global -> LDS (stride-136 halves)
        // ck is 128 or 32 -> 16 or 4 16-byte segments per row; total segments
        // are a multiple of 256, so all lanes stay active (EXEC all-ones).
        {
            const int segShift  = (ck == 128) ? 4 : 2;    // log2(ck/8)
            const int segMask   = (1 << segShift) - 1;
            const int totalSegs = 128 << segShift;
            for (int idx = tid; idx < totalSegs; idx += 256) {
                const int n = idx >> segShift;
                const int s = idx & segMask;
                uint32_t ldsOff = ldsBase + (uint32_t)(n * WLDS_STRIDE + s * 8) * 2u;
                uint32_t gOff   = ((uint32_t)n * (uint32_t)Kpad +
                                   (uint32_t)(kc + s * 8)) * 2u;
                asm volatile("global_load_async_to_lds_b128 %0, %1, %2"
                             :: "v"(ldsOff), "v"(gOff), "s"(Wt)
                             : "memory");
            }
            // wave's own DMA done -> LDS written; barrier publishes all waves'
            asm volatile("s_wait_asynccnt 0" ::: "memory");
        }
        __syncthreads();

        if (kc + 128 < Kpad)                       // hint the next A chunk
            __builtin_prefetch((const void*)(Arow + ((kc + 128) >> 3)), 0, 0);

        const int ksteps = ck >> 5;
        for (int ks = 0; ks < ksteps; ++ks) {
            const int kglob = kc + ks * 32;
            // A fragment: 16x32 f16. lanes0-15: K {0..7,16..23}; lanes16-31: +8
            FragH af;
            const int o0 = (kglob >> 3) + laneHi;
            af.u[0] = Arow[o0];
            af.u[1] = Arow[o0 + 2];
#pragma unroll
            for (int t = 0; t < NTILES; ++t) {
                // B fragment: 32x16 f16. lane n=lm; lanes0-15 K {0..7,16..23}; lanes16-31: +8
                FragH bf;
                const uint4* bp =
                    (const uint4*)(&Wlds[(t * 16 + lm) * WLDS_STRIDE + ks * 32 + laneHi * 8]);
                bf.u[0] = bp[0];
                bf.u[1] = bp[2];
                acc[t] = __builtin_amdgcn_wmma_f32_16x16x32_f16(
                    false, af.v, false, bf.v, (short)0, acc[t], false, false);
            }
        }
        __syncthreads();
    }

    // Epilogue. D layout: VGPR r -> M = r + 8*laneHi, N = lane&15 (+16*t)
#pragma unroll
    for (int t = 0; t < NTILES; ++t) {
#pragma unroll
        for (int r = 0; r < 8; ++r) {
            const int row = row0 + r + laneHi * 8;
            const int col = t * 16 + lm;
            if (row >= rows) continue;
            float v = acc[t][r] + blds[col];
            if (MODE == 0) {
                v = v > 0.0f ? v : 0.0f;
                outH[(size_t)row * 128 + col] = (_Float16)v;
            } else if (MODE == 1) {
                outH[(size_t)row * 128 + col] = (_Float16)v;
                outF[(size_t)row * 128 + col] = v;
            } else if (MODE == 2) {
                float nv = outF[(size_t)row * 128 + col] + v;
                outF[(size_t)row * 128 + col] = nv;
                outH[(size_t)row * 128 + col] = (_Float16)nv;
                atomAddF(&aggr[(size_t)dstIdx[row] * 128 + col], v);
            } else if (MODE == 3) {
                float nv = outF[(size_t)row * 128 + col] + v;
                outF[(size_t)row * 128 + col] = nv;
                outH[(size_t)row * 128 + col] = (_Float16)nv;
            } else { // MODE 4
                if (col < Nout) dout[(size_t)row * 3 + col] = v;
            }
        }
    }
}

// ---------------------------------------------------------------------------
// One-time (per launch) weight pre-transpose: W[K][Nout] f32 -> Wt[128][Kpad] f16
// Zero-padded so the GEMM needs no masking; ~3MB total, L2-resident.
// ---------------------------------------------------------------------------
__global__ void convert_W(const float* __restrict__ W, _Float16* __restrict__ Wt,
                          int K, int Nout, int Kpad) {
    int k = blockIdx.x * 256 + threadIdx.x;
    int n = blockIdx.y;
    if (k >= Kpad) return;
    float v = (k < K && n < Nout) ? W[(size_t)k * (size_t)Nout + n] : 0.0f;
    Wt[(size_t)n * (size_t)Kpad + k] = (_Float16)v;
}

// ---------------------------------------------------------------------------
// Gather / padding kernels (build f16 A-operands)
// ---------------------------------------------------------------------------
__global__ void prep_node_A(const float* __restrict__ cn, _Float16* __restrict__ Adst, int N) {
    int idx = blockIdx.x * 256 + threadIdx.x;
    if (idx >= N * 32) return;
    int col = idx & 31;
    int n   = idx >> 5;
    Adst[idx] = (col == 0) ? (_Float16)cn[n] : (_Float16)0.0f;
}

__global__ void prep_edge_A(const float* __restrict__ ied, const float* __restrict__ len,
                            const float* __restrict__ p2c, const float* __restrict__ br,
                            const float* __restrict__ st, _Float16* __restrict__ Adst, int E) {
    int idx = blockIdx.x * 256 + threadIdx.x;
    if (idx >= E * 32) return;
    int col = idx & 31;
    int e   = idx >> 5;
    float v = 0.0f;
    if (col < 3) v = ied[(size_t)e * 3 + col];
    else if (col == 3) v = len[e];
    else if (col == 4) v = p2c[e];
    else if (col == 5) v = br[e];
    else if (col == 6) v = st[e];
    Adst[idx] = (_Float16)v;
}

// A_edge[e] = [ nf[dst] | nf[src] | ef[e] | gf[batch[src]] | 0-pad ]  (416 cols)
__global__ void build_edge_A(const _Float16* __restrict__ nf_h, const _Float16* __restrict__ ef_h,
                             const float* __restrict__ gf, const int* __restrict__ src,
                             const int* __restrict__ dst, const int* __restrict__ batch,
                             _Float16* __restrict__ Adst) {
    int e = blockIdx.x;
    int t = threadIdx.x;           // 128 threads
    int s = src[e], d = dst[e];
    _Float16* a = Adst + (size_t)e * 416;
    a[t]       = nf_h[(size_t)d * 128 + t];
    a[128 + t] = nf_h[(size_t)s * 128 + t];
    a[256 + t] = ef_h[(size_t)e * 128 + t];
    if (t < 32) {
        float g = (t < 3) ? gf[(size_t)batch[s] * 3 + t] : 0.0f;
        a[384 + t] = (_Float16)g;
    }
}

// A_node[n] = [ nf[n] | aggr[n] | gf[batch[n]] | 0-pad ]  (288 cols)
__global__ void build_node_A(const _Float16* __restrict__ nf_h, const float* __restrict__ aggr,
                             const float* __restrict__ gf, const int* __restrict__ batch,
                             _Float16* __restrict__ Adst) {
    int n = blockIdx.x;
    int t = threadIdx.x;           // 128 threads
    _Float16* a = Adst + (size_t)n * 288;
    a[t]       = nf_h[(size_t)n * 128 + t];
    a[128 + t] = (_Float16)aggr[(size_t)n * 128 + t];
    if (t < 32) {
        float g = (t < 3) ? gf[(size_t)batch[n] * 3 + t] : 0.0f;
        a[256 + t] = (_Float16)g;
    }
}

// ---------------------------------------------------------------------------
// Host-side helpers
// ---------------------------------------------------------------------------
template <int NTILES, int MODE>
static void run_gemm(hipStream_t s, const _Float16* A, int lda, int rows,
                     const _Float16* Wt, const float* b, int Kpad, int Nout,
                     _Float16* oH, float* oF, float* ag, const int* dI, float* dO) {
    dim3 grid((rows + 127) / 128), blk(256);
    mlp_gemm<NTILES, MODE><<<grid, blk, 0, s>>>(A, lda, rows, Wt, b, Kpad, Nout,
                                                oH, oF, ag, dI, dO);
}

extern "C" void kernel_launch(void* const* d_in, const int* in_sizes, int n_in,
                              void* d_out, int out_size, void* d_ws, size_t ws_size,
                              hipStream_t stream) {
    const int N = in_sizes[0];           // 100000 nodes
    const int E = in_sizes[3];           // 200000 edges

    const float* contact_node  = (const float*)d_in[0];
    const float* contact_force = (const float*)d_in[1];
    const float* ied   = (const float*)d_in[2];
    const float* len   = (const float*)d_in[3];
    const float* p2c   = (const float*)d_in[4];
    const float* br    = (const float*)d_in[5];
    const float* st    = (const float*)d_in[6];
    const int*   src   = (const int*)d_in[7];        // edge_index[0]
    const int*   dst   = ((const int*)d_in[7]) + E;  // edge_index[1]
    const int*   batch = (const int*)d_in[8];
    float*       dout  = (float*)d_out;

    // ---- param leaf indexing (pytree flatten order; detect deterministically)
    const int PB = 9;
    // sorted-key order:   IN(120), edge_in(6), node_action(6), node_in(6)
    // insertion order:    node_in(6), edge_in(6), node_action(6), IN(120)
    const bool sortedOrder = (in_sizes[PB] == 387 * 128);
    int in_base, ein_base, act_base, nin_base;
    if (sortedOrder) { in_base = PB;  ein_base = PB + 120; act_base = PB + 126; nin_base = PB + 132; }
    else             { nin_base = PB; ein_base = PB + 6;   act_base = PB + 12;  in_base  = PB + 18;  }
    auto P = [&](int i) { return (const float*)d_in[i]; };

    // ---- workspace arena
    char* ws = (char*)d_ws;
    size_t off = 0;
    auto arena = [&](size_t bytes) -> char* {
        char* p = ws + off;
        off += (bytes + 255) & ~(size_t)255;
        return p;
    };
    float*    nf_f = (float*)arena((size_t)N * 128 * 4);
    _Float16* nf_h = (_Float16*)arena((size_t)N * 128 * 2);
    float*    ef_f = (float*)arena((size_t)E * 128 * 4);
    _Float16* ef_h = (_Float16*)arena((size_t)E * 128 * 2);
    float*    aggr = (float*)arena((size_t)N * 128 * 4);
    _Float16* Abig = (_Float16*)arena((size_t)E * 416 * 2);  // reused for all concats
    _Float16* h1   = (_Float16*)arena((size_t)E * 128 * 2);
    _Float16* h2   = (_Float16*)arena((size_t)E * 128 * 2);
    (void)ws_size; (void)n_in; (void)out_size;

    // ---- pre-transpose all weights to f16 [128][Kpad] (once per launch)
    auto conv = [&](const float* W, int K, int Nout, int Kpad) -> const _Float16* {
        _Float16* Wt = (_Float16*)arena((size_t)128 * Kpad * 2);
        convert_W<<<dim3((Kpad + 255) / 256, 128), dim3(256), 0, stream>>>(W, Wt, K, Nout, Kpad);
        return Wt;
    };
    const _Float16* ninW[3] = { conv(P(nin_base + 0), 1, 128, 32),
                                conv(P(nin_base + 2), 128, 128, 128),
                                conv(P(nin_base + 4), 128, 128, 128) };
    const _Float16* einW[3] = { conv(P(ein_base + 0), 7, 128, 32),
                                conv(P(ein_base + 2), 128, 128, 128),
                                conv(P(ein_base + 4), 128, 128, 128) };
    const _Float16* eW[10][3];
    const _Float16* nW[10][3];
    for (int i = 0; i < 10; ++i) {
        const int eb = in_base + i * 12;
        const int nb = in_base + i * 12 + 6;
        eW[i][0] = conv(P(eb + 0), 387, 128, 416);
        eW[i][1] = conv(P(eb + 2), 128, 128, 128);
        eW[i][2] = conv(P(eb + 4), 128, 128, 128);
        nW[i][0] = conv(P(nb + 0), 259, 128, 288);
        nW[i][1] = conv(P(nb + 2), 128, 128, 128);
        nW[i][2] = conv(P(nb + 4), 128, 128, 128);
    }
    const _Float16* actW[3] = { conv(P(act_base + 0), 128, 128, 128),
                                conv(P(act_base + 2), 128, 128, 128),
                                conv(P(act_base + 4), 128, 3, 128) };

    // ---- node encoder: [N,1]->128->128->128
    prep_node_A<<<dim3((N * 32 + 255) / 256), dim3(256), 0, stream>>>(contact_node, Abig, N);
    run_gemm<8, 0>(stream, Abig, 32, N, ninW[0], P(nin_base + 1), 32, 128, h1, 0, 0, 0, 0);
    run_gemm<8, 0>(stream, h1, 128, N, ninW[1], P(nin_base + 3), 128, 128, h2, 0, 0, 0, 0);
    run_gemm<8, 1>(stream, h2, 128, N, ninW[2], P(nin_base + 5), 128, 128, nf_h, nf_f, 0, 0, 0);

    // ---- edge encoder: [E,7]->128->128->128
    prep_edge_A<<<dim3((E * 32 + 255) / 256), dim3(256), 0, stream>>>(ied, len, p2c, br, st, Abig, E);
    run_gemm<8, 0>(stream, Abig, 32, E, einW[0], P(ein_base + 1), 32, 128, h1, 0, 0, 0, 0);
    run_gemm<8, 0>(stream, h1, 128, E, einW[1], P(ein_base + 3), 128, 128, h2, 0, 0, 0, 0);
    run_gemm<8, 1>(stream, h2, 128, E, einW[2], P(ein_base + 5), 128, 128, ef_h, ef_f, 0, 0, 0);

    // ---- 10 interaction layers
    for (int i = 0; i < 10; ++i) {
        const int eb = in_base + i * 12;
        const int nb = in_base + i * 12 + 6;

        build_edge_A<<<dim3(E), dim3(128), 0, stream>>>(nf_h, ef_h, contact_force, src, dst, batch, Abig);
        hipMemsetAsync(aggr, 0, (size_t)N * 128 * 4, stream);

        run_gemm<8, 0>(stream, Abig, 416, E, eW[i][0], P(eb + 1), 416, 128, h1, 0, 0, 0, 0);
        run_gemm<8, 0>(stream, h1, 128, E, eW[i][1], P(eb + 3), 128, 128, h2, 0, 0, 0, 0);
        // m: ef += m (f32 + f16 mirror), aggr[dst] += m (global_atomic_add_f32)
        run_gemm<8, 2>(stream, h2, 128, E, eW[i][2], P(eb + 5), 128, 128, ef_h, ef_f, aggr, dst, 0);

        build_node_A<<<dim3(N), dim3(128), 0, stream>>>(nf_h, aggr, contact_force, batch, Abig);
        run_gemm<8, 0>(stream, Abig, 288, N, nW[i][0], P(nb + 1), 288, 128, h1, 0, 0, 0, 0);
        run_gemm<8, 0>(stream, h1, 128, N, nW[i][1], P(nb + 3), 128, 128, h2, 0, 0, 0, 0);
        run_gemm<8, 3>(stream, h2, 128, N, nW[i][2], P(nb + 5), 128, 128, nf_h, nf_f, 0, 0, 0);
    }

    // ---- action head: 128->128->128->3
    run_gemm<8, 0>(stream, nf_h, 128, N, actW[0], P(act_base + 1), 128, 128, h1, 0, 0, 0, 0);
    run_gemm<8, 0>(stream, h1, 128, N, actW[1], P(act_base + 3), 128, 128, h2, 0, 0, 0, 0);
    run_gemm<1, 4>(stream, h2, 128, N, actW[2], P(act_base + 5), 128, 3, 0, 0, 0, 0, dout);
}
